// Dualalign_32409823216126
// MI455X (gfx1250) — matrix operand order repaired and checked by reading, compiled, not run on hardware
//
#include <hip/hip_runtime.h>

#define FEAT 320
#define NPTS 2048
#define BATCH 32

typedef __attribute__((ext_vector_type(16))) __bf16 v16bf;
typedef __attribute__((ext_vector_type(8)))  __bf16 v8bf;
typedef __attribute__((ext_vector_type(8)))  float  v8f;

// ---------------------------------------------------------------------------
// WMMA core
// ---------------------------------------------------------------------------
static __device__ __forceinline__ v8f wmma_bf16(v16bf a, v16bf b, v8f c) {
    // D = A x B + C, f32 accumulate. args: neg_a, A, neg_b, B, c_mod, C, reuse_a, reuse_b
    return __builtin_amdgcn_wmma_f32_16x16x32_bf16(false, a, false, b, (short)0, c,
                                                   false, false);
}

// ---------------------------------------------------------------------------
// Fragment loaders following CDNA5 WMMA VGPR layouts (CDNA5 ISA)
//
// A 16x32 (MxK): lane L(0-15): row M=L, K=0..7 & 16..23; lane 16-31: row L-16,
//                K=8..15 & 24..31.
// B 32x16 (KxN) stored as B^T rows: lane L: col N=L%16, lanes 0-15 K=0..7 & 16..23,
//                lanes 16-31 K=8..15 & 24..31 (same as A).
// ---------------------------------------------------------------------------

// From global memory, row-major with leading dim lda
static __device__ __forceinline__ v16bf load_a_frag(const __bf16* __restrict__ A,
                                                    int lda, int row0, int k0) {
    const int lane = threadIdx.x & 31;
    const int r    = row0 + (lane & 15);
    const int kb   = k0 + ((lane >> 4) << 3);
    const __bf16* p = A + (size_t)r * lda + kb;
    v8bf lo = *(const v8bf*)p;
    v8bf hi = *(const v8bf*)(p + 16);
    v16bf f;
#pragma unroll
    for (int j = 0; j < 8; ++j) { f[j] = lo[j]; f[j + 8] = hi[j]; }
    return f;
}

static __device__ __forceinline__ v16bf load_b_frag(const __bf16* __restrict__ Bt,
                                                    int ldb, int k0, int col0) {
    const int lane = threadIdx.x & 31;
    const int n    = col0 + (lane & 15);
    const int kk   = k0 + ((lane >> 4) << 3);
    const __bf16* p = Bt + (size_t)n * ldb + kk;
    v8bf lo = *(const v8bf*)p;
    v8bf hi = *(const v8bf*)(p + 16);
    v16bf f;
#pragma unroll
    for (int j = 0; j < 8; ++j) { f[j] = lo[j]; f[j + 8] = hi[j]; }
    return f;
}

// From LDS tiles padded to 40 bf16 per row (80 B: keeps 16-B alignment, spreads banks)
static __device__ __forceinline__ v16bf lds_a_frag(const __bf16* As, int row0) {
    const int lane = threadIdx.x & 31;
    const int r    = row0 + (lane & 15);
    const int kb   = (lane >> 4) << 3;
    const __bf16* p = As + r * 40 + kb;
    v8bf lo = *(const v8bf*)p;
    v8bf hi = *(const v8bf*)(p + 16);
    v16bf f;
#pragma unroll
    for (int j = 0; j < 8; ++j) { f[j] = lo[j]; f[j + 8] = hi[j]; }
    return f;
}

static __device__ __forceinline__ v16bf lds_b_frag(const __bf16* Bs, int col0) {
    const int lane = threadIdx.x & 31;
    const int n    = col0 + (lane & 15);
    const int kk   = (lane >> 4) << 3;
    const __bf16* p = Bs + n * 40 + kk;
    v8bf lo = *(const v8bf*)p;
    v8bf hi = *(const v8bf*)(p + 16);
    v16bf f;
#pragma unroll
    for (int j = 0; j < 8; ++j) { f[j] = lo[j]; f[j + 8] = hi[j]; }
    return f;
}

// ---------------------------------------------------------------------------
// f32 -> bf16 conversion
// ---------------------------------------------------------------------------
__global__ void cvt_f32_bf16_kernel(const float* __restrict__ in,
                                    __bf16* __restrict__ out, size_t n) {
    size_t i      = (size_t)blockIdx.x * blockDim.x + threadIdx.x;
    size_t stride = (size_t)gridDim.x * blockDim.x;
    for (; i < n; i += stride) out[i] = (__bf16)in[i];
}

// ---------------------------------------------------------------------------
// Projection GEMM, LDS-blocked: C[o,f] = sum_i W[o,i] * X[b,f,i]
// Macro-tile 128(M=o) x 64(N=f) per 8-wave block; wave = 32x32 (2x2 WMMA tiles).
// K-step 32: stage A 128x32 and B 32x64 in LDS -> 42.7 FLOP/B from global.
// transposed=1 -> store [b,FEAT,NPTS] (q,k); 0 -> [b,NPTS,FEAT] (v)
// ---------------------------------------------------------------------------
__global__ void proj_gemm_kernel(const __bf16* __restrict__ W,
                                 const __bf16* __restrict__ X,
                                 __bf16* __restrict__ Out,
                                 int transposed) {
    __shared__ __bf16 As[128][40];   // [m][k]
    __shared__ __bf16 Bs[64][40];    // [n][k]  (B^T storage)

    const int blk = blockIdx.x;                // BATCH * 16 * 5
    const int b   = blk / (16 * 5);
    const int rr  = blk % (16 * 5);
    const int mB  = (rr / 5) * 128;            // o base
    const int nB  = (rr % 5) * 64;             // f base
    const __bf16* Xb = X + (size_t)b * FEAT * NPTS;

    const int t   = threadIdx.x;
    const int wv  = t >> 5;
    const int m0w = (wv >> 1) * 32;            // wave offset in macro-tile
    const int n0w = (wv & 1) * 32;

    const int ar = t >> 1, akp = (t & 1) * 16; // A stage: 128 rows x 32 k
    const int br = t >> 2, bkp = (t & 3) * 8;  // B stage: 64 rows x 32 k

    v8f acc[2][2] = {};

    for (int k0 = 0; k0 < NPTS; k0 += 32) {
        __syncthreads();
        const __bf16* ga = W + (size_t)(mB + ar) * NPTS + k0 + akp;
        *(v8bf*)&As[ar][akp]     = *(const v8bf*)ga;
        *(v8bf*)&As[ar][akp + 8] = *(const v8bf*)(ga + 8);
        *(v8bf*)&Bs[br][bkp] =
            *(const v8bf*)(Xb + (size_t)(nB + br) * NPTS + k0 + bkp);
        __syncthreads();

        v16bf a0 = lds_a_frag(&As[0][0], m0w);
        v16bf a1 = lds_a_frag(&As[0][0], m0w + 16);
        v16bf b0 = lds_b_frag(&Bs[0][0], n0w);
        v16bf b1 = lds_b_frag(&Bs[0][0], n0w + 16);
        acc[0][0] = wmma_bf16(a0, b0, acc[0][0]);
        acc[0][1] = wmma_bf16(a0, b1, acc[0][1]);
        acc[1][0] = wmma_bf16(a1, b0, acc[1][0]);
        acc[1][1] = wmma_bf16(a1, b1, acc[1][1]);
    }

    const int lane = t & 31;
    const int nn = lane & 15;                  // column within 16x16 tile (f)
    const int mb = (lane >> 4) << 3;           // first of lane's 8 rows (o)
#pragma unroll
    for (int ms = 0; ms < 2; ++ms) {
#pragma unroll
        for (int ns = 0; ns < 2; ++ns) {
            const int oo = mB + m0w + 16 * ms + mb;
            const int ff = nB + n0w + 16 * ns + nn;
            if (transposed) {
                v8bf h;
#pragma unroll
                for (int j = 0; j < 8; ++j) h[j] = (__bf16)acc[ms][ns][j];
                *(v8bf*)(Out + (size_t)b * FEAT * NPTS + (size_t)ff * NPTS + oo) = h;
            } else {
                const size_t base = (size_t)b * NPTS * FEAT;
#pragma unroll
                for (int j = 0; j < 8; ++j)
                    Out[base + (size_t)(oo + j) * FEAT + ff] = (__bf16)acc[ms][ns][j];
            }
        }
    }
}

// ---------------------------------------------------------------------------
// Attention + softmax fused:
// attn[b,l,m] = softmax_m( sum_o qT[b,l,o]*kT[b,m,o] / sqrt(FEAT) )
// 20 waves/block; block covers rows l0..l0+15, wave = one m-tile. Softmax over
// full 320-wide rows via LDS + wave32 shfl reductions.
// ---------------------------------------------------------------------------
__global__ void attn_softmax_kernel(const __bf16* __restrict__ qT,
                                    const __bf16* __restrict__ kT,
                                    __bf16* __restrict__ attn) {
    __shared__ float sm[16][FEAT + 8];
    const int b    = blockIdx.x / 20;
    const int l0   = (blockIdx.x % 20) * 16;
    const int wv   = threadIdx.x >> 5;
    const int lane = threadIdx.x & 31;
    const int m0   = wv * 16;

    const __bf16* A = qT + (size_t)b * FEAT * NPTS;
    const __bf16* K = kT + (size_t)b * FEAT * NPTS;

    v8f acc = {};
    for (int k0 = 0; k0 < NPTS; k0 += 32) {
        v16bf a  = load_a_frag(A, NPTS, l0, k0);
        v16bf bb = load_b_frag(K, NPTS, k0, m0);
        acc = wmma_bf16(a, bb, acc);
    }

    const float invTemp = 0.05590169943749474f;   // 1/sqrt(320)
    const int n  = lane & 15;
    const int mb = (lane >> 4) << 3;
#pragma unroll
    for (int j = 0; j < 8; ++j) sm[mb + j][m0 + n] = acc[j] * invTemp;
    __syncthreads();

    if (wv < 16) {                                // wave wv owns row l0+wv
        float v[10];
        float mx = -1e30f;
#pragma unroll
        for (int i = 0; i < 10; ++i) { v[i] = sm[wv][lane + 32 * i]; mx = fmaxf(mx, v[i]); }
#pragma unroll
        for (int off = 16; off > 0; off >>= 1) mx = fmaxf(mx, __shfl_xor(mx, off, 32));
        float s = 0.f;
#pragma unroll
        for (int i = 0; i < 10; ++i) { v[i] = __expf(v[i] - mx); s += v[i]; }
#pragma unroll
        for (int off = 16; off > 0; off >>= 1) s += __shfl_xor(s, off, 32);
        const float inv = 1.f / s;
        __bf16* dst = attn + (size_t)b * FEAT * FEAT + (size_t)(l0 + wv) * FEAT;
#pragma unroll
        for (int i = 0; i < 10; ++i) dst[lane + 32 * i] = (__bf16)(v[i] * inv);
    }
}

// ---------------------------------------------------------------------------
// Output GEMM + residual, LDS-blocked:
// out[b,l,o] = sum_m attn[b,l,m] * v[b,o,m] + query[b,l,o]
// Macro-tile 64(M=l) x 128(N=o); wave = 32x32; K-step 32 over FEAT=320.
// ---------------------------------------------------------------------------
__global__ void out_gemm_residual_kernel(const __bf16* __restrict__ attn,
                                         const __bf16* __restrict__ V,
                                         const float* __restrict__ query,
                                         float* __restrict__ out) {
    __shared__ __bf16 As[64][40];    // [l][k]
    __shared__ __bf16 Bs[128][40];   // [o][k]  (B^T storage)

    const int blk = blockIdx.x;                 // BATCH * 5 * 16
    const int b   = blk / (5 * 16);
    const int rr  = blk % (5 * 16);
    const int mB  = (rr / 16) * 64;             // l base
    const int nB  = (rr % 16) * 128;            // o base
    const __bf16* Ab = attn + (size_t)b * FEAT * FEAT;
    const __bf16* Vb = V    + (size_t)b * NPTS * FEAT;

    const int t   = threadIdx.x;
    const int wv  = t >> 5;
    const int m0w = (wv >> 2) * 32;
    const int n0w = (wv & 3) * 32;

    const int ar = t >> 2, akp = (t & 3) * 8;   // A stage: 64 rows x 32 k
    const int br = t >> 1, bkp = (t & 1) * 16;  // B stage: 128 rows x 32 k

    v8f acc[2][2] = {};

    for (int k0 = 0; k0 < FEAT; k0 += 32) {
        __syncthreads();
        *(v8bf*)&As[ar][akp] =
            *(const v8bf*)(Ab + (size_t)(mB + ar) * FEAT + k0 + akp);
        const __bf16* gb = Vb + (size_t)(nB + br) * FEAT + k0 + bkp;
        *(v8bf*)&Bs[br][bkp]     = *(const v8bf*)gb;
        *(v8bf*)&Bs[br][bkp + 8] = *(const v8bf*)(gb + 8);
        __syncthreads();

        v16bf a0 = lds_a_frag(&As[0][0], m0w);
        v16bf a1 = lds_a_frag(&As[0][0], m0w + 16);
        v16bf b0 = lds_b_frag(&Bs[0][0], n0w);
        v16bf b1 = lds_b_frag(&Bs[0][0], n0w + 16);
        acc[0][0] = wmma_bf16(a0, b0, acc[0][0]);
        acc[0][1] = wmma_bf16(a0, b1, acc[0][1]);
        acc[1][0] = wmma_bf16(a1, b0, acc[1][0]);
        acc[1][1] = wmma_bf16(a1, b1, acc[1][1]);
    }

    const int lane = t & 31;
    const int nn = lane & 15;
    const int mb = (lane >> 4) << 3;
    const size_t base = (size_t)b * FEAT * NPTS;
#pragma unroll
    for (int ms = 0; ms < 2; ++ms) {
#pragma unroll
        for (int ns = 0; ns < 2; ++ns) {
            const int ll = mB + m0w + 16 * ms + mb;
            const int oo = nB + n0w + 16 * ns + nn;
#pragma unroll
            for (int j = 0; j < 8; ++j) {
                const size_t idx = base + (size_t)(ll + j) * NPTS + oo;
                out[idx] = acc[ms][ns][j] + query[idx];
            }
        }
    }
}

// ---------------------------------------------------------------------------
// LayerNorm over FEAT axis, layout [b, f, n]: pass 1 per-(b,n) stats
// (coalesced over n), pass 2 normalize in place.
// ---------------------------------------------------------------------------
__global__ void ln_stats_kernel(const float* __restrict__ out,
                                float* __restrict__ mu, float* __restrict__ rstd) {
    const int gid = blockIdx.x * blockDim.x + threadIdx.x;   // [0, B*NPTS)
    const int b = gid / NPTS, n = gid % NPTS;
    const float* p = out + (size_t)b * FEAT * NPTS + n;
    float s = 0.f, ss = 0.f;
    for (int l = 0; l < FEAT; ++l) {
        float v = p[(size_t)l * NPTS];
        s += v; ss += v * v;
    }
    const float m   = s * (1.0f / FEAT);
    const float var = ss * (1.0f / FEAT) - m * m;
    mu[gid]   = m;
    rstd[gid] = rsqrtf(var + 1e-5f);
}

__global__ void ln_apply_kernel(float* __restrict__ out, const float* __restrict__ mu,
                                const float* __restrict__ rstd,
                                const float* __restrict__ gamma,
                                const float* __restrict__ beta) {
    const size_t idx = (size_t)blockIdx.x * blockDim.x + threadIdx.x;
    const int b  = (int)(idx / ((size_t)FEAT * NPTS));
    const size_t rem = idx % ((size_t)FEAT * NPTS);
    const int l  = (int)(rem / NPTS);
    const int n  = (int)(rem % NPTS);
    const int col = b * NPTS + n;
    out[idx] = (out[idx] - mu[col]) * rstd[col] * gamma[l] + beta[l];
}

// ---------------------------------------------------------------------------
// Host-side launch
// ---------------------------------------------------------------------------
extern "C" void kernel_launch(void* const* d_in, const int* in_sizes, int n_in,
                              void* d_out, int out_size, void* d_ws, size_t ws_size,
                              hipStream_t stream) {
    (void)in_sizes; (void)n_in; (void)out_size; (void)ws_size;
    const float* query   = (const float*)d_in[0];  // [32,320,2048]
    const float* support = (const float*)d_in[1];  // [32,320,2048]
    const float* Wq      = (const float*)d_in[2];  // [2048,2048]
    const float* Wk      = (const float*)d_in[3];
    const float* Wv      = (const float*)d_in[4];
    const float* gamma   = (const float*)d_in[5];  // [320]
    const float* beta    = (const float*)d_in[6];  // [320]
    float* out = (float*)d_out;                    // [32,320,2048] f32

    const size_t W_ELEMS = (size_t)NPTS * NPTS;
    const size_t X_ELEMS = (size_t)BATCH * FEAT * NPTS;
    const size_t A_ELEMS = (size_t)BATCH * FEAT * FEAT;

    char* ws = (char*)d_ws;
    __bf16* Wq_bf = (__bf16*)ws;  ws += W_ELEMS * sizeof(__bf16);
    __bf16* Wk_bf = (__bf16*)ws;  ws += W_ELEMS * sizeof(__bf16);
    __bf16* Wv_bf = (__bf16*)ws;  ws += W_ELEMS * sizeof(__bf16);
    __bf16* s_bf  = (__bf16*)ws;  ws += X_ELEMS * sizeof(__bf16);
    __bf16* x_bf  = (__bf16*)ws;  ws += X_ELEMS * sizeof(__bf16);
    __bf16* qT    = (__bf16*)ws;  ws += X_ELEMS * sizeof(__bf16);   // [b,FEAT,NPTS]
    __bf16* kT    = (__bf16*)ws;  ws += X_ELEMS * sizeof(__bf16);   // [b,FEAT,NPTS]
    __bf16* v_d   = (__bf16*)ws;  ws += X_ELEMS * sizeof(__bf16);   // [b,NPTS,FEAT]
    __bf16* at_bf = (__bf16*)ws;  ws += A_ELEMS * sizeof(__bf16);
    float*  mu    = (float*)ws;   ws += (size_t)BATCH * NPTS * sizeof(float);
    float*  rstd  = (float*)ws;

    // 1) f32 -> bf16
    auto cvt = [&](const float* src, __bf16* dst, size_t n) {
        int blocks = (int)((n + 1023) / 1024);
        cvt_f32_bf16_kernel<<<blocks, 256, 0, stream>>>(src, dst, n);
    };
    cvt(Wq, Wq_bf, W_ELEMS);
    cvt(Wk, Wk_bf, W_ELEMS);
    cvt(Wv, Wv_bf, W_ELEMS);
    cvt(support, s_bf, X_ELEMS);
    cvt(query,   x_bf, X_ELEMS);

    // 2) projections (LDS-blocked GEMM): q = Wq x support (transposed store),
    //    k = Wk x query (transposed), v = Wv x query (direct [b,o,f])
    const int proj_blocks = BATCH * 16 * 5;    // 2560 blocks, 8 waves each
    proj_gemm_kernel<<<proj_blocks, 256, 0, stream>>>(Wq_bf, s_bf, qT, 1);
    proj_gemm_kernel<<<proj_blocks, 256, 0, stream>>>(Wk_bf, x_bf, kT, 1);
    proj_gemm_kernel<<<proj_blocks, 256, 0, stream>>>(Wv_bf, x_bf, v_d, 0);

    // 3) attention + softmax (fused), bf16 result
    attn_softmax_kernel<<<BATCH * 20, 640, 0, stream>>>(qT, kT, at_bf);

    // 4) attn x v^T + residual -> d_out (pre-LN, f32), LDS-blocked
    const int out_blocks = BATCH * 5 * 16;     // 2560 blocks
    out_gemm_residual_kernel<<<out_blocks, 256, 0, stream>>>(at_bf, v_d, query, out);

    // 5) LayerNorm over FEAT in place
    ln_stats_kernel<<<(BATCH * NPTS) / 256, 256, 0, stream>>>(out, mu, rstd);
    ln_apply_kernel<<<(int)(X_ELEMS / 256), 256, 0, stream>>>(out, mu, rstd, gamma, beta);
}
